// CNNEMLStageNet_3332894621846
// MI455X (gfx1250) — matrix operand order repaired and checked by reading, not compiled, not run on hardware
//
#include <hip/hip_runtime.h>
#include <hip/hip_bf16.h>
#include <math.h>

// ---------------------------------------------------------------------------
// CNNEMLStageNet for MI455X (gfx1250, wave32, WMMA).
// Decomposed edge-MLP: per-pixel GEMMs (bf16 WMMA, f32 acc) + per-edge gate.
// Round 3: hoist B (weight) WMMA fragments out of the row-tile loop — each
// wave's column tile is fixed, so the 4 k-step B fragments are loaded from
// LDS once and reused across all 4 row tiles (halves hot-loop ds traffic).
// ---------------------------------------------------------------------------

typedef __attribute__((ext_vector_type(16))) __bf16 bf16x16;
typedef __attribute__((ext_vector_type(8)))  __bf16 bf16x8;
typedef __attribute__((ext_vector_type(4)))  __bf16 bf16x4;
typedef __attribute__((ext_vector_type(8)))  float  f32x8;

#define NPIX   36864      // B*H*W = 4*96*96
#define CDIM   128
#define HH     96
#define WW     96
#define HWPIX  9216       // 96*96
#define AST    136        // LDS A-tile stride (bf16 elems): 16B-aligned rows
#define WST    136        // LDS W^T-tile stride (bf16 elems)
#define RSTF   136        // LDS f32 result-tile stride: 16B-aligned rows

__device__ __forceinline__ float wave_sum32(float v) {
    #pragma unroll
    for (int off = 16; off > 0; off >>= 1) v += __shfl_xor(v, off, 32);
    return v;
}

__device__ __forceinline__ float softplus_f(float x) {
    return (x > 20.f) ? x : log1pf(__expf(x));
}

__device__ __forceinline__ float gelu_exact(float x) {
    return 0.5f * x * (1.f + erff(x * 0.70710678118654752f));
}

// Load the four k-step B fragments for this wave's fixed column tile.
// wT: LDS holding W^T (wT[c*WST + k]).  Lanes 0-15 = cols; K = K0 + {half*8+0..7, 16+half*8+0..7}.
__device__ __forceinline__ void load_bfrag(const __bf16* wT, int colTile,
                                           int lane, bf16x16* bv) {
    const int ncol = lane & 15;
    const int half = lane >> 4;
    #pragma unroll
    for (int kk = 0; kk < 4; ++kk) {
        const __bf16* bp = wT + (colTile * 16 + ncol) * WST + kk * 32 + half * 8;
        bf16x8 blo = *(const bf16x8*)(bp);
        bf16x8 bhi = *(const bf16x8*)(bp + 16);
        #pragma unroll
        for (int e = 0; e < 8; ++e) { bv[kk][e] = blo[e]; bv[kk][e + 8] = bhi[e]; }
    }
}

// One 16x16 f32 tile of Out = A[16x128] @ W[128x16] using pre-loaded B frags.
__device__ __forceinline__ f32x8 wmma_rowtile(const __bf16* aT, const bf16x16* bv,
                                              int rowTile, int lane) {
    const int mrow = lane & 15;
    const int half = lane >> 4;
    f32x8 acc = {0.f, 0.f, 0.f, 0.f, 0.f, 0.f, 0.f, 0.f};
    #pragma unroll
    for (int kk = 0; kk < 4; ++kk) {
        // A fragment: lanes 0-15 = rows; K = K0 + {half*8+0..7, 16+half*8+0..7}
        const __bf16* ap = aT + (rowTile * 16 + mrow) * AST + kk * 32 + half * 8;
        bf16x8 alo = *(const bf16x8*)(ap);
        bf16x8 ahi = *(const bf16x8*)(ap + 16);
        bf16x16 av;
        #pragma unroll
        for (int e = 0; e < 8; ++e) { av[e] = alo[e]; av[e + 8] = ahi[e]; }
        acc = __builtin_amdgcn_wmma_f32_16x16x32_bf16(
                  false, av, false, bv[kk], (short)0, acc, false, false);
    }
    return acc;
}

// One [64x128] @ [128x128] GEMM with LDS-staged epilogue and coalesced stores.
__device__ __forceinline__ void gemm_block(
    const __bf16* __restrict__ aT, __bf16* __restrict__ wT,
    float* __restrict__ resL, const __bf16* __restrict__ WT /* transposed */,
    const float* __restrict__ biasPtr, float* __restrict__ Out,
    int rowBase, int wave, int lane, int tid) {
    __syncthreads();
    // stage W^T: straight vectorized copy (128 rows x 128 cols of bf16)
    for (int i = tid; i < 128 * 16; i += 256) {
        int c = i >> 4, kv = (i & 15) * 8;
        *(bf16x8*)(wT + c * WST + kv) = *(const bf16x8*)(WT + c * CDIM + kv);
    }
    __syncthreads();
    bf16x16 bv[4];
    load_bfrag(wT, wave, lane, bv);
    const int col = wave * 16 + (lane & 15);
    const float bias = biasPtr ? biasPtr[col] : 0.f;
    #pragma unroll
    for (int rt = 0; rt < 4; ++rt) {
        f32x8 acc = wmma_rowtile(aT, bv, rt, lane);
        #pragma unroll
        for (int j = 0; j < 8; ++j) {
            int lr = rt * 16 + j + 8 * (lane >> 4);
            resL[lr * RSTF + col] = acc[j] + bias;
        }
    }
    __syncthreads();
    // coalesced vector store of the 64x128 f32 tile
    for (int i = tid; i < 64 * 32; i += 256) {
        int r = i >> 5, cv = (i & 31) * 4;
        *(float4*)(Out + (size_t)(rowBase + r) * CDIM + cv) =
            *(const float4*)(resL + r * RSTF + cv);
    }
}

// ---------------------------------------------------------------------------
// Kernel 1: weight prep.  Fold the (center - neighbor) block of W1 into the
// center/neighbor blocks; write ALL matrices TRANSPOSED (W^T[n][k]) in bf16 so
// GEMM staging is a pure vector copy.  Build rel-pos+b1 tables.
// ---------------------------------------------------------------------------
__global__ __launch_bounds__(256) void prep_kernel(
    const float* __restrict__ d_w1, const float* __restrict__ d_b1,
    const float* __restrict__ r_w1, const float* __restrict__ r_b1,
    const float* __restrict__ v_w,  const float* __restrict__ o_w,
    const float* __restrict__ rel_pos,
    __bf16* cwDcT, __bf16* cwDnT, __bf16* cwRcT, __bf16* cwRnT,
    __bf16* vWT, __bf16* oWT, float* rpcD, float* rpcR) {
    int idx = blockIdx.x * 256 + threadIdx.x;
    if (idx < CDIM * CDIM) {
        int k = idx >> 7, h = idx & 127;    // h = output column
        int tidx = h * CDIM + k;            // transposed destination
        float dc = d_w1[k * CDIM + h], dn = d_w1[(128 + k) * CDIM + h],
              dd = d_w1[(256 + k) * CDIM + h];
        cwDcT[tidx] = (__bf16)(dc + dd);
        cwDnT[tidx] = (__bf16)(dn - dd);
        float rc = r_w1[k * CDIM + h], rn = r_w1[(128 + k) * CDIM + h],
              rd = r_w1[(256 + k) * CDIM + h];
        cwRcT[tidx] = (__bf16)(rc + rd);
        cwRnT[tidx] = (__bf16)(rn - rd);
        vWT[tidx] = (__bf16)v_w[idx];
        oWT[tidx] = (__bf16)o_w[idx];
    }
    if (idx < 9 * CDIM) {
        int nb = idx >> 7, h = idx & 127;
        float sd = d_b1[h], sr = r_b1[h];
        #pragma unroll
        for (int r = 0; r < 8; ++r) {
            float rp = rel_pos[nb * 8 + r];
            sd += rp * d_w1[(384 + r) * CDIM + h];
            sr += rp * r_w1[(384 + r) * CDIM + h];
        }
        rpcD[idx] = sd;
        rpcR[idx] = sr;
    }
}

// ---------------------------------------------------------------------------
// Kernel 2: LayerNorm1 -> bf16 X.  One wave per row (128 channels / 32 lanes).
// ---------------------------------------------------------------------------
__global__ __launch_bounds__(256) void ln1_kernel(
    const float* __restrict__ tokens, const float* __restrict__ ln1w,
    const float* __restrict__ ln1b, __bf16* __restrict__ X) {
    int row  = blockIdx.x * 8 + (threadIdx.x >> 5);
    int lane = threadIdx.x & 31;
    const float4 x = *(const float4*)(tokens + (size_t)row * CDIM + lane * 4);
    float s = wave_sum32(x.x + x.y + x.z + x.w);
    float mean = s * (1.f / 128.f);
    float dx0 = x.x - mean, dx1 = x.y - mean, dx2 = x.z - mean, dx3 = x.w - mean;
    float sq = wave_sum32(dx0 * dx0 + dx1 * dx1 + dx2 * dx2 + dx3 * dx3);
    float rs = rsqrtf(sq * (1.f / 128.f) + 1e-5f);
    const float4 w = *(const float4*)(ln1w + lane * 4);
    const float4 b = *(const float4*)(ln1b + lane * 4);
    bf16x4 o;
    o[0] = (__bf16)(dx0 * rs * w.x + b.x);
    o[1] = (__bf16)(dx1 * rs * w.y + b.y);
    o[2] = (__bf16)(dx2 * rs * w.z + b.z);
    o[3] = (__bf16)(dx3 * rs * w.w + b.w);
    *(bf16x4*)(X + (size_t)row * CDIM + lane * 4) = o;
}

// ---------------------------------------------------------------------------
// Kernel 3: five per-pixel GEMMs sharing one staged A tile (64 rows of X).
//   Ad = X@(Wc_d+Wd_d), Bd = X@(Wn_d-Wd_d), Ar, Br, V = X@v_w + v_b
// ---------------------------------------------------------------------------
__global__ __launch_bounds__(256) void gemm5_kernel(
    const __bf16* __restrict__ X,
    const __bf16* __restrict__ w0T, const __bf16* __restrict__ w1T,
    const __bf16* __restrict__ w2T, const __bf16* __restrict__ w3T,
    const __bf16* __restrict__ w4T, const float* __restrict__ v_b,
    float* __restrict__ Ad, float* __restrict__ Bd,
    float* __restrict__ Ar, float* __restrict__ Br, float* __restrict__ V) {
    __shared__ __align__(16) __bf16 aT[64 * AST];
    __shared__ __align__(16) __bf16 wT[128 * WST];
    __shared__ __align__(16) float  resL[64 * RSTF];
    const int tid  = threadIdx.x;
    const int wave = tid >> 5;
    const int lane = tid & 31;
    const int rowBase = blockIdx.x * 64;

    // stage A: 64 x 128 bf16, 16B vectors
    for (int i = tid; i < 64 * 16; i += 256) {
        int r = i >> 4, kv = (i & 15) * 8;
        *(bf16x8*)(aT + r * AST + kv) =
            *(const bf16x8*)(X + (size_t)(rowBase + r) * CDIM + kv);
    }

    gemm_block(aT, wT, resL, w0T, nullptr, Ad, rowBase, wave, lane, tid);
    gemm_block(aT, wT, resL, w1T, nullptr, Bd, rowBase, wave, lane, tid);
    gemm_block(aT, wT, resL, w2T, nullptr, Ar, rowBase, wave, lane, tid);
    gemm_block(aT, wT, resL, w3T, nullptr, Br, rowBase, wave, lane, tid);
    gemm_block(aT, wT, resL, w4T, v_b,     V,  rowBase, wave, lane, tid);
}

// ---------------------------------------------------------------------------
// Kernel 4: per-pixel edge gate + gated message.  One wave per pixel.
// drive/resistance = w2 . gelu(A[center] + B[neighbor] + rpc[nb]), gate=sig(EML)
// OOB neighbor (zero-padded norm): B contribution = 0, value = v_b exactly.
// ---------------------------------------------------------------------------
__global__ __launch_bounds__(256) void edge_kernel(
    const float* __restrict__ Ad, const float* __restrict__ Bd,
    const float* __restrict__ Ar, const float* __restrict__ Br,
    const float* __restrict__ V,  const float* __restrict__ v_b,
    const float* __restrict__ rpcD, const float* __restrict__ rpcR,
    const float* __restrict__ d_w2, const float* __restrict__ d_b2,
    const float* __restrict__ r_w2, const float* __restrict__ r_b2,
    const float* __restrict__ g_raw, const float* __restrict__ g_lam,
    const float* __restrict__ g_bias, __bf16* __restrict__ MSG) {
    const int p    = blockIdx.x * 8 + (threadIdx.x >> 5);
    const int lane = threadIdx.x & 31;
    const int cb   = lane * 4;
    const int bimg = p / HWPIX;
    const int rem  = p - bimg * HWPIX;
    const int y = rem / WW, x = rem - (rem / WW) * WW;

    // Prefetch the three neighbor rows of the stencil (global_prefetch_b8).
    #pragma unroll
    for (int dyy = 0; dyy < 3; ++dyy) {
        int ny = y + dyy - 1;
        if ((unsigned)ny < HH) {
            size_t q0 = (size_t)(bimg * HWPIX + ny * WW + x) * CDIM + cb;
            __builtin_prefetch(Bd + q0, 0, 1);
            __builtin_prefetch(Br + q0, 0, 1);
            __builtin_prefetch(V  + q0, 0, 1);
        }
    }

    const float4 ad  = *(const float4*)(Ad + (size_t)p * CDIM + cb);
    const float4 ar  = *(const float4*)(Ar + (size_t)p * CDIM + cb);
    const float4 w2d = *(const float4*)(d_w2 + cb);
    const float4 w2r = *(const float4*)(r_w2 + cb);
    const float4 vbv = *(const float4*)(v_b + cb);
    const float gamma = softplus_f(g_raw[0]);
    const float lam = g_lam[0], ebias = g_bias[0];
    const float b2d = d_b2[0], b2r = r_b2[0];

    float m0 = 0.f, m1 = 0.f, m2 = 0.f, m3 = 0.f, gm = 0.f;

    #pragma unroll
    for (int k = 0; k < 9; ++k) {
        const int dy = k / 3 - 1, dx = k % 3 - 1;
        const int ny = y + dy, nx = x + dx;
        const bool in = ((unsigned)ny < HH) && ((unsigned)nx < WW);
        const int q = in ? (bimg * HWPIX + ny * WW + nx) : 0;
        float4 bd = {0.f, 0.f, 0.f, 0.f}, br = {0.f, 0.f, 0.f, 0.f};
        float4 vv = vbv;
        if (in) {
            bd = *(const float4*)(Bd + (size_t)q * CDIM + cb);
            br = *(const float4*)(Br + (size_t)q * CDIM + cb);
            vv = *(const float4*)(V  + (size_t)q * CDIM + cb);
        }
        const float4 rd = *(const float4*)(rpcD + k * CDIM + cb);
        const float4 rr = *(const float4*)(rpcR + k * CDIM + cb);

        float pd = gelu_exact(ad.x + bd.x + rd.x) * w2d.x
                 + gelu_exact(ad.y + bd.y + rd.y) * w2d.y
                 + gelu_exact(ad.z + bd.z + rd.z) * w2d.z
                 + gelu_exact(ad.w + bd.w + rd.w) * w2d.w;
        float pr = gelu_exact(ar.x + br.x + rr.x) * w2r.x
                 + gelu_exact(ar.y + br.y + rr.y) * w2r.y
                 + gelu_exact(ar.z + br.z + rr.z) * w2r.z
                 + gelu_exact(ar.w + br.w + rr.w) * w2r.w;
        const float drive = wave_sum32(pd) + b2d;
        const float resis = wave_sum32(pr) + b2r;
        const float cond = drive / (softplus_f(resis) + gamma + 1e-6f);
        const float e = fminf(fmaxf(lam * cond, -3.f), 3.f) + ebias;
        const float gate = 1.f / (1.f + __expf(-e));
        m0 += gate * vv.x; m1 += gate * vv.y; m2 += gate * vv.z; m3 += gate * vv.w;
        gm += gate;
    }
    const float inv = 1.f / fmaxf(gm, 1e-6f);
    bf16x4 o;
    o[0] = (__bf16)(m0 * inv); o[1] = (__bf16)(m1 * inv);
    o[2] = (__bf16)(m2 * inv); o[3] = (__bf16)(m3 * inv);
    *(bf16x4*)(MSG + (size_t)p * CDIM + cb) = o;
}

// ---------------------------------------------------------------------------
// Kernel 5: update = MSG @ o_w + o_b + tokens, then LayerNorm2 -> out (f32).
// GEMM epilogue stages rows in LDS so the full 128-channel row is local.
// ---------------------------------------------------------------------------
__global__ __launch_bounds__(256) void outln_kernel(
    const __bf16* __restrict__ MSG, const __bf16* __restrict__ oWT,
    const float* __restrict__ o_b, const float* __restrict__ tokens,
    const float* __restrict__ ln2w, const float* __restrict__ ln2b,
    float* __restrict__ out) {
    __shared__ __align__(16) __bf16 aT[64 * AST];
    __shared__ __align__(16) __bf16 wT[128 * WST];
    __shared__ __align__(16) float  resL[64 * RSTF];
    const int tid  = threadIdx.x;
    const int wave = tid >> 5;
    const int lane = tid & 31;
    const int rowBase = blockIdx.x * 64;

    for (int i = tid; i < 64 * 16; i += 256) {
        int r = i >> 4, kv = (i & 15) * 8;
        *(bf16x8*)(aT + r * AST + kv) =
            *(const bf16x8*)(MSG + (size_t)(rowBase + r) * CDIM + kv);
    }
    for (int i = tid; i < 128 * 16; i += 256) {
        int c = i >> 4, kv = (i & 15) * 8;
        *(bf16x8*)(wT + c * WST + kv) = *(const bf16x8*)(oWT + c * CDIM + kv);
    }
    __syncthreads();

    bf16x16 bv[4];
    load_bfrag(wT, wave, lane, bv);
    const int col = wave * 16 + (lane & 15);
    const float ob = o_b[col];
    #pragma unroll
    for (int rt = 0; rt < 4; ++rt) {
        f32x8 acc = wmma_rowtile(aT, bv, rt, lane);
        #pragma unroll
        for (int j = 0; j < 8; ++j) {
            int lr = rt * 16 + j + 8 * (lane >> 4);
            resL[lr * RSTF + col] =
                acc[j] + ob + tokens[(size_t)(rowBase + lr) * CDIM + col];
        }
    }
    __syncthreads();

    // LayerNorm2: each wave handles 8 rows; lane covers channels lane*4..+3.
    #pragma unroll
    for (int rr = 0; rr < 8; ++rr) {
        const int lr = wave * 8 + rr;
        const float4 v = *(const float4*)(resL + lr * RSTF + lane * 4);
        float mean = wave_sum32(v.x + v.y + v.z + v.w) * (1.f / 128.f);
        float d0 = v.x - mean, d1 = v.y - mean, d2 = v.z - mean, d3 = v.w - mean;
        float var = wave_sum32(d0 * d0 + d1 * d1 + d2 * d2 + d3 * d3) * (1.f / 128.f);
        float rs = rsqrtf(var + 1e-5f);
        const float4 w = *(const float4*)(ln2w + lane * 4);
        const float4 b = *(const float4*)(ln2b + lane * 4);
        float4 o;
        o.x = d0 * rs * w.x + b.x;
        o.y = d1 * rs * w.y + b.y;
        o.z = d2 * rs * w.z + b.z;
        o.w = d3 * rs * w.w + b.w;
        *(float4*)(out + (size_t)(rowBase + lr) * CDIM + lane * 4) = o;
    }
}

// ---------------------------------------------------------------------------
extern "C" void kernel_launch(void* const* d_in, const int* in_sizes, int n_in,
                              void* d_out, int out_size, void* d_ws, size_t ws_size,
                              hipStream_t stream) {
    const float* tokens  = (const float*)d_in[0];
    const float* ln1_w   = (const float*)d_in[1];
    const float* ln1_b   = (const float*)d_in[2];
    const float* ln2_w   = (const float*)d_in[3];
    const float* ln2_b   = (const float*)d_in[4];
    const float* rel_pos = (const float*)d_in[5];
    const float* d_w1    = (const float*)d_in[6];
    const float* d_b1    = (const float*)d_in[7];
    const float* d_w2    = (const float*)d_in[8];
    const float* d_b2    = (const float*)d_in[9];
    const float* r_w1    = (const float*)d_in[10];
    const float* r_b1    = (const float*)d_in[11];
    const float* r_w2    = (const float*)d_in[12];
    const float* r_b2    = (const float*)d_in[13];
    const float* v_w     = (const float*)d_in[14];
    const float* v_b     = (const float*)d_in[15];
    const float* o_w     = (const float*)d_in[16];
    const float* o_b     = (const float*)d_in[17];
    const float* eml_g   = (const float*)d_in[18];
    const float* eml_lam = (const float*)d_in[19];
    const float* eml_bias= (const float*)d_in[20];

    char* ws = (char*)d_ws;
    const size_t Nc2 = (size_t)NPIX * CDIM * 2;   // bf16 plane
    const size_t Nc4 = (size_t)NPIX * CDIM * 4;   // f32 plane
    size_t off = 0;
    __bf16* X    = (__bf16*)(ws + off); off += Nc2;
    float*  Ad   = (float*)(ws + off);  off += Nc4;
    float*  Bd   = (float*)(ws + off);  off += Nc4;
    float*  Ar   = (float*)(ws + off);  off += Nc4;
    float*  Br   = (float*)(ws + off);  off += Nc4;
    float*  V    = (float*)(ws + off);  off += Nc4;
    __bf16* MSG  = (__bf16*)(ws + off); off += Nc2;
    __bf16* cwDcT = (__bf16*)(ws + off); off += 128 * 128 * 2;
    __bf16* cwDnT = (__bf16*)(ws + off); off += 128 * 128 * 2;
    __bf16* cwRcT = (__bf16*)(ws + off); off += 128 * 128 * 2;
    __bf16* cwRnT = (__bf16*)(ws + off); off += 128 * 128 * 2;
    __bf16* vWT   = (__bf16*)(ws + off); off += 128 * 128 * 2;
    __bf16* oWT   = (__bf16*)(ws + off); off += 128 * 128 * 2;
    float*  rpcD  = (float*)(ws + off);  off += 9 * 128 * 4;
    float*  rpcR  = (float*)(ws + off);  off += 9 * 128 * 4;

    prep_kernel<<<64, 256, 0, stream>>>(d_w1, d_b1, r_w1, r_b1, v_w, o_w, rel_pos,
                                        cwDcT, cwDnT, cwRcT, cwRnT, vWT, oWT,
                                        rpcD, rpcR);
    ln1_kernel<<<NPIX / 8, 256, 0, stream>>>(tokens, ln1_w, ln1_b, X);
    gemm5_kernel<<<NPIX / 64, 256, 0, stream>>>(X, cwDcT, cwDnT, cwRcT, cwRnT, vWT,
                                                v_b, Ad, Bd, Ar, Br, V);
    edge_kernel<<<NPIX / 8, 256, 0, stream>>>(Ad, Bd, Ar, Br, V, v_b, rpcD, rpcR,
                                              d_w2, d_b2, r_w2, r_b2,
                                              eml_g, eml_lam, eml_bias, MSG);
    outln_kernel<<<NPIX / 64, 256, 0, stream>>>(MSG, oWT, o_b, tokens, ln2_w, ln2_b,
                                                (float*)d_out);
}
